// SRNetLayer_80805514707498
// MI455X (gfx1250) — matrix operand order repaired and checked by reading, not compiled, not run on hardware
//
#include <hip/hip_runtime.h>

// ---------------------------------------------------------------------------
// Problem constants: L=2, H=8, D=64, HID=512, FF=2048, S=256, B=8, IN=512.
// ---------------------------------------------------------------------------
#define LNUM 2
#define HNUM 8
#define DDIM 64
#define HID 512
#define FFD 2048
#define SEQ 256
#define BSZ 8
#define MROWS 2048

typedef __bf16 bf16_t;
typedef bf16_t v16bf __attribute__((ext_vector_type(16)));
typedef bf16_t v8bf  __attribute__((ext_vector_type(8)));
typedef bf16_t v4bf  __attribute__((ext_vector_type(4)));
typedef float  v8f   __attribute__((ext_vector_type(8)));

__device__ __forceinline__ bf16_t f2bf(float f) {
  unsigned u = __builtin_bit_cast(unsigned, f);
  unsigned r = u + 0x7FFFu + ((u >> 16) & 1u);   // round-to-nearest-even
  unsigned short hs = (unsigned short)(r >> 16);
  return __builtin_bit_cast(bf16_t, hs);
}

enum { EPI_BIAS = 0, EPI_RES = 1, EPI_BIAS_RELU = 2, EPI_BIAS_RES = 3 };

// ---------------------------------------------------------------------------
// f32 -> bf16 pack (4 elements/thread, b128 in / b64 out). n % 1024 == 0.
// ---------------------------------------------------------------------------
__global__ __launch_bounds__(256) void cvt_f32_bf16(
    const float* __restrict__ in, bf16_t* __restrict__ out, int n) {
  const int idx = (blockIdx.x * 256 + threadIdx.x) * 4;
  const float4 v = *(const float4*)(in + idx);
  v4bf o;
  o[0] = f2bf(v.x); o[1] = f2bf(v.y); o[2] = f2bf(v.z); o[3] = f2bf(v.w);
  *(v4bf*)(out + idx) = o;
}

// ---------------------------------------------------------------------------
// WMMA GEMM: C[M,N] = A[M,K] @ W[N,K]^T (+ bias / residual / relu).
// A, W pre-packed bf16. One wave computes a 32(M) x 64(N) strip:
// 2 A fragments x 4 B fragments -> 8 v_wmma_f32_16x16x32_bf16 per K-step
// on 12 global_load_b128 (B fragments shared by both M tiles).
// Block = 128 threads = 4 waves stacked in M covering 128(M) x 64(N).
// Grid: (M/128)*(N/64) blocks.
// ---------------------------------------------------------------------------
template <int EPI, bool OUTBF>
__global__ __launch_bounds__(128) void gemm_wmma(
    const bf16_t* __restrict__ A, const bf16_t* __restrict__ W,
    const float* __restrict__ bias, const float* __restrict__ res,
    void* __restrict__ Cv, int M, int N, int K) {
  const int lane = threadIdx.x & 31;
  const int wave = threadIdx.x >> 5;   // 0..3
  const int half = lane >> 4;          // 0: lanes 0-15, 1: lanes 16-31
  const int mr   = lane & 15;

  const int bn_cnt = N >> 6;
  const int bm   = blockIdx.x / bn_cnt;
  const int bn   = blockIdx.x - bm * bn_cnt;
  const int row0 = bm * 128 + wave * 32;
  const int col0 = bn * 64;

  const bf16_t* __restrict__ arow0 = A + (size_t)(row0 + mr) * K;
  const bf16_t* __restrict__ arow1 = A + (size_t)(row0 + 16 + mr) * K;
  const bf16_t* brow[4];
#pragma unroll
  for (int nt = 0; nt < 4; ++nt)
    brow[nt] = W + (size_t)(col0 + nt * 16 + mr) * K + 8 * half;

  v8f acc[2][4];
#pragma unroll
  for (int mt = 0; mt < 2; ++mt)
#pragma unroll
    for (int nt = 0; nt < 4; ++nt)
#pragma unroll
      for (int r = 0; r < 8; ++r) acc[mt][nt][r] = 0.f;

  for (int k0 = 0; k0 < K; k0 += 32) {
    // Prefetch next K slab (global_prefetch_b8).
    __builtin_prefetch(arow0 + k0 + 32, 0, 1);
    __builtin_prefetch(arow1 + k0 + 32, 0, 1);
    __builtin_prefetch(brow[0] + k0 + 32, 0, 1);

    // A fragments (ISA 7.12.2 16-bit A 16x32): per lane two contiguous
    // 8-element chunks at K offsets 8*half and 16+8*half.
    const v8bf a00 = *(const v8bf*)(arow0 + k0 + 8 * half);
    const v8bf a01 = *(const v8bf*)(arow0 + k0 + 16 + 8 * half);
    const v8bf a10 = *(const v8bf*)(arow1 + k0 + 8 * half);
    const v8bf a11 = *(const v8bf*)(arow1 + k0 + 16 + 8 * half);
    v16bf af0, af1;
#pragma unroll
    for (int e = 0; e < 8; ++e) {
      af0[e] = a00[e]; af0[8 + e] = a01[e];
      af1[e] = a10[e]; af1[8 + e] = a11[e];
    }

#pragma unroll
    for (int nt = 0; nt < 4; ++nt) {
      // B fragment: lane holds column col0+nt*16+mr, two 8-element K chunks
      // at k0 + 8*half (folded into brow) and k0 + 16 + 8*half. Shared by both M tiles.
      const v8bf b0 = *(const v8bf*)(brow[nt] + k0);
      const v8bf b1 = *(const v8bf*)(brow[nt] + k0 + 16);
      v16bf bfv;
#pragma unroll
      for (int e = 0; e < 8; ++e) { bfv[e] = b0[e]; bfv[8 + e] = b1[e]; }
      acc[0][nt] = __builtin_amdgcn_wmma_f32_16x16x32_bf16(
          false, af0, false, bfv, (short)0, acc[0][nt], false, false);
      acc[1][nt] = __builtin_amdgcn_wmma_f32_16x16x32_bf16(
          false, af1, false, bfv, (short)0, acc[1][nt], false, false);
    }
  }

  // Epilogue. C/D layout: VGPR r -> M = r + 8*half, N = mr (per 16x16 tile).
#pragma unroll
  for (int nt = 0; nt < 4; ++nt) {
    const int n = col0 + nt * 16 + mr;
    float bcol = 0.f;
    if (EPI == EPI_BIAS || EPI == EPI_BIAS_RELU || EPI == EPI_BIAS_RES)
      bcol = bias[n];
#pragma unroll
    for (int mt = 0; mt < 2; ++mt) {
#pragma unroll
      for (int r = 0; r < 8; ++r) {
        const int m = row0 + mt * 16 + r + 8 * half;
        float v = acc[mt][nt][r] + bcol;
        if (EPI == EPI_BIAS_RELU) v = v > 0.f ? v : 0.f;
        if (EPI == EPI_RES || EPI == EPI_BIAS_RES) v += res[(size_t)m * N + n];
        if (OUTBF)
          ((bf16_t*)Cv)[(size_t)m * N + n] = f2bf(v);
        else
          ((float*)Cv)[(size_t)m * N + n] = v;
      }
    }
  }
}

// ---------------------------------------------------------------------------
// Per-head softmax: in (S,B,HID) -> xs (B,H,S,D), softmax over D=64.
// One wave per row; 8 rows per 256-thread block.
// ---------------------------------------------------------------------------
__global__ __launch_bounds__(256) void softmax_heads(
    const float* __restrict__ in, float* __restrict__ xs) {
  const int row  = blockIdx.x * 8 + (threadIdx.x >> 5);  // (s*B+b)*H + h
  const int lane = threadIdx.x & 31;
  const int h  = row & (HNUM - 1);
  const int sb = row >> 3;          // s*B + b
  const int b  = sb & (BSZ - 1);
  const int s  = sb >> 3;

  const float* p = in + (size_t)sb * HID + h * DDIM;
  float v0 = p[lane], v1 = p[lane + 32];
  float m = fmaxf(v0, v1);
#pragma unroll
  for (int off = 16; off; off >>= 1) m = fmaxf(m, __shfl_xor(m, off, 32));
  float e0 = __expf(v0 - m), e1 = __expf(v1 - m);
  float sum = e0 + e1;
#pragma unroll
  for (int off = 16; off; off >>= 1) sum += __shfl_xor(sum, off, 32);
  const float inv = 1.0f / sum;
  float* q = xs + (((size_t)(b * HNUM + h) * SEQ + s) * DDIM);
  q[lane]      = e0 * inv;
  q[lane + 32] = e1 * inv;
}

// ---------------------------------------------------------------------------
// LayerNorm over rows of HID=512; emits bf16 (only consumer is the FF1 GEMM).
// ---------------------------------------------------------------------------
__global__ __launch_bounds__(256) void layernorm_rows(
    const float* __restrict__ in, const float* __restrict__ g,
    const float* __restrict__ bb, bf16_t* __restrict__ out) {
  const int row  = blockIdx.x * 8 + (threadIdx.x >> 5);
  const int lane = threadIdx.x & 31;
  const float* p = in + (size_t)row * HID;
  float v[16];
  float sum = 0.f;
#pragma unroll
  for (int j = 0; j < 16; ++j) { v[j] = p[lane + 32 * j]; sum += v[j]; }
#pragma unroll
  for (int off = 16; off; off >>= 1) sum += __shfl_xor(sum, off, 32);
  const float mu = sum * (1.0f / HID);
  float var = 0.f;
#pragma unroll
  for (int j = 0; j < 16; ++j) { float t = v[j] - mu; var += t * t; }
#pragma unroll
  for (int off = 16; off; off >>= 1) var += __shfl_xor(var, off, 32);
  const float r = rsqrtf(var * (1.0f / HID) + 1e-5f);
  bf16_t* o = out + (size_t)row * HID;
#pragma unroll
  for (int j = 0; j < 16; ++j) {
    const int c = lane + 32 * j;
    o[c] = f2bf((v[j] - mu) * r * g[c] + bb[c]);
  }
}

// ---------------------------------------------------------------------------
// SRWM sequential scan. One 64-thread block per (b,h); states in LDS (49KB of
// the 320KB WGP pool). Thread i owns row i of Wy/Wq/Wk/wb -> updates are
// row-local. f32 throughout (serial recurrence: keep exact precision).
// y output emitted as bf16 (only consumer is the Wo GEMM).
// ---------------------------------------------------------------------------
__global__ __launch_bounds__(64) void srwm_scan(
    const float* __restrict__ xs,                       // (B,H,S,D)
    const float* __restrict__ Wy_s, const float* __restrict__ Wq_s,
    const float* __restrict__ Wk_s, const float* __restrict__ wb_s,
    const float* __restrict__ Wy_p, const float* __restrict__ Wq_p,
    const float* __restrict__ Wk_p, const float* __restrict__ wb_p,
    bf16_t* __restrict__ ys,                            // (S,B,HID) bf16
    float* __restrict__ sy, float* __restrict__ sq,
    float* __restrict__ sk, float* __restrict__ sb, int l) {
  __shared__ float Wy[DDIM * DDIM], Wq[DDIM * DDIM], Wk[DDIM * DDIM];
  __shared__ float wb[DDIM * 4];
  __shared__ float xv[DDIM], qv[DDIM], kv[DDIM], dv[DDIM];
  __shared__ float betas[4], dbs[4];

  const int bh = blockIdx.x;          // b*H + h
  const int b  = bh >> 3;
  const int h  = bh & (HNUM - 1);
  const int i  = threadIdx.x;         // 0..63

  const size_t so  = ((size_t)((l * BSZ + b) * HNUM + h)) * (DDIM * DDIM);
  const size_t po  = ((size_t)(l * HNUM + h)) * (DDIM * DDIM);
  const size_t sob = ((size_t)((l * BSZ + b) * HNUM + h)) * (DDIM * 4);
  const size_t pob = ((size_t)(l * HNUM + h)) * (DDIM * 4);

  for (int j = 0; j < DDIM; ++j) {
    Wy[i * DDIM + j] = Wy_s[so + i * DDIM + j] + Wy_p[po + i * DDIM + j];
    Wq[i * DDIM + j] = Wq_s[so + i * DDIM + j] + Wq_p[po + i * DDIM + j];
    Wk[i * DDIM + j] = Wk_s[so + i * DDIM + j] + Wk_p[po + i * DDIM + j];
  }
  for (int c = 0; c < 4; ++c)
    wb[i * 4 + c] = wb_s[sob + i * 4 + c] + wb_p[pob + i * 4 + c];
  __syncthreads();

  const float* xrow = xs + (size_t)bh * SEQ * DDIM;

  for (int t = 0; t < SEQ; ++t) {
    xv[i] = xrow[t * DDIM + i];
    __syncthreads();

    float y = 0.f, qd = 0.f, kd = 0.f;
    for (int j = 0; j < DDIM; ++j) {
      const float xj = xv[j];
      y  += Wy[i * DDIM + j] * xj;
      qd += Wq[i * DDIM + j] * xj;
      kd += Wk[i * DDIM + j] * xj;
    }
    qv[i] = qd; kv[i] = kd;
    __syncthreads();
    float mq = -3.4e38f, mk = -3.4e38f;
    for (int j = 0; j < DDIM; ++j) {
      mq = fmaxf(mq, qv[j]);
      mk = fmaxf(mk, kv[j]);
    }
    const float eq = __expf(qd - mq), ek = __expf(kd - mk);
    __syncthreads();
    qv[i] = eq; kv[i] = ek;
    __syncthreads();
    float sqs = 0.f, sks = 0.f;
    for (int j = 0; j < DDIM; ++j) { sqs += qv[j]; sks += kv[j]; }
    const float qi = eq / sqs, ki = ek / sks;
    const float di = qi - ki;
    __syncthreads();
    kv[i] = ki; dv[i] = di;
    __syncthreads();

    if (i < 4) {
      float bacc = 0.f, dacc = 0.f;
      for (int j = 0; j < DDIM; ++j) {
        bacc += wb[j * 4 + i] * xv[j];
        dacc += wb[j * 4 + i] * dv[j];
      }
      betas[i] = 1.0f / (1.0f + __expf(-bacc));
      dbs[i]   = dacc;
    }

    float dy = 0.f, dq = 0.f, dk = 0.f;
    for (int j = 0; j < DDIM; ++j) {
      const float dj = dv[j];
      dy += Wy[i * DDIM + j] * dj;
      dq += Wq[i * DDIM + j] * dj;
      dk += Wk[i * DDIM + j] * dj;
    }
    __syncthreads();  // betas/dbs visible; wb reads done before wb update

    const float b0 = betas[0] * dy, b1 = betas[1] * dq, b2 = betas[2] * dk;
    for (int j = 0; j < DDIM; ++j) {
      const float kj = kv[j];
      Wy[i * DDIM + j] += b0 * kj;
      Wq[i * DDIM + j] += b1 * kj;
      Wk[i * DDIM + j] += b2 * kj;
    }
    const float b3k = betas[3] * kv[i];
    for (int c = 0; c < 4; ++c) wb[i * 4 + c] += b3k * dbs[c];

    ys[((size_t)t * BSZ + b) * HID + h * DDIM + i] = f2bf(y);
    __syncthreads();
  }

  for (int j = 0; j < DDIM; ++j) {
    sy[so + i * DDIM + j] = Wy[i * DDIM + j] - Wy_p[po + i * DDIM + j];
    sq[so + i * DDIM + j] = Wq[i * DDIM + j] - Wq_p[po + i * DDIM + j];
    sk[so + i * DDIM + j] = Wk[i * DDIM + j] - Wk_p[po + i * DDIM + j];
  }
  for (int c = 0; c < 4; ++c)
    sb[sob + i * 4 + c] = wb[i * 4 + c] - wb_p[pob + i * 4 + c];
}

// ---------------------------------------------------------------------------
// Orchestration
// ---------------------------------------------------------------------------
extern "C" void kernel_launch(void* const* d_in, const int* in_sizes, int n_in,
                              void* d_out, int out_size, void* d_ws,
                              size_t ws_size, hipStream_t stream) {
  const float* x    = (const float*)d_in[0];
  const float* WyS  = (const float*)d_in[1];
  const float* WqS  = (const float*)d_in[2];
  const float* WkS  = (const float*)d_in[3];
  const float* wbS  = (const float*)d_in[4];
  const float* inW  = (const float*)d_in[5];
  const float* inB  = (const float*)d_in[6];
  const float* WyP  = (const float*)d_in[7];
  const float* WqP  = (const float*)d_in[8];
  const float* WkP  = (const float*)d_in[9];
  const float* wbP  = (const float*)d_in[10];
  const float* Wo   = (const float*)d_in[11];
  const float* fW1  = (const float*)d_in[12];
  const float* fb1  = (const float*)d_in[13];
  const float* fW2  = (const float*)d_in[14];
  const float* fb2  = (const float*)d_in[15];
  const float* lnG  = (const float*)d_in[16];
  const float* lnB  = (const float*)d_in[17];

  float* out = (float*)d_out;
  float* sy  = out + (size_t)MROWS * HID;
  float* sq  = sy + (size_t)LNUM * BSZ * HNUM * DDIM * DDIM;
  float* sk  = sq + (size_t)LNUM * BSZ * HNUM * DDIM * DDIM;
  float* sb  = sk + (size_t)LNUM * BSZ * HNUM * DDIM * DDIM;

  const size_t ACT = (size_t)MROWS * HID;   // 1048576 elements
  float*  bufA = (float*)d_ws;              // f32 activations ping
  float*  bufB = bufA + ACT;                // f32 activations pong
  float*  xsB  = bufB + ACT;                // f32 (B,H,S,D) softmaxed heads
  bf16_t* xbf  = (bf16_t*)(xsB + ACT);      // bf16 input activations
  bf16_t* ysbf = xbf + ACT;                 // bf16 scan output (S,B,HID)
  bf16_t* lnbf = ysbf + ACT;                // bf16 layernorm output
  bf16_t* ffbf = lnbf + ACT;                // bf16 FF hidden (2048,2048)
  bf16_t* wInb = ffbf + (size_t)MROWS * FFD;            // in_W   bf16
  bf16_t* wOob = wInb + (size_t)HID * HID;              // Wo     bf16 (L)
  bf16_t* wF1b = wOob + (size_t)LNUM * HID * HID;       // ff_W1  bf16 (L)
  bf16_t* wF2b = wF1b + (size_t)LNUM * FFD * HID;       // ff_W2  bf16 (L)

  const dim3 blk(256);
  const dim3 gblk(128);   // GEMM blocks: 4 waves

  // One-shot bf16 packing of GEMM operands that aren't produced on-device.
  cvt_f32_bf16<<<(MROWS * HID) / 1024, blk, 0, stream>>>(x, xbf, MROWS * HID);
  cvt_f32_bf16<<<(HID * HID) / 1024, blk, 0, stream>>>(inW, wInb, HID * HID);
  cvt_f32_bf16<<<(LNUM * HID * HID) / 1024, blk, 0, stream>>>(
      Wo, wOob, LNUM * HID * HID);
  cvt_f32_bf16<<<(LNUM * FFD * HID) / 1024, blk, 0, stream>>>(
      fW1, wF1b, LNUM * FFD * HID);
  cvt_f32_bf16<<<(LNUM * HID * FFD) / 1024, blk, 0, stream>>>(
      fW2, wF2b, LNUM * HID * FFD);

  // out0 = x @ in_W^T + in_b : grid (2048/128)*(512/64) = 128 blocks
  gemm_wmma<EPI_BIAS, false><<<128, gblk, 0, stream>>>(
      xbf, wInb, inB, nullptr, bufA, MROWS, HID, HID);

  float* cur = bufA;
  for (int l = 0; l < LNUM; ++l) {
    softmax_heads<<<(MROWS * HNUM) / 8, blk, 0, stream>>>(cur, xsB);
    srwm_scan<<<BSZ * HNUM, dim3(64), 0, stream>>>(
        xsB, WyS, WqS, WkS, wbS, WyP, WqP, WkP, wbP, ysbf, sy, sq, sk, sb, l);
    // out = h + y @ Wo^T
    gemm_wmma<EPI_RES, false><<<128, gblk, 0, stream>>>(
        ysbf, wOob + (size_t)l * HID * HID, nullptr, cur, bufB,
        MROWS, HID, HID);
    layernorm_rows<<<MROWS / 8, blk, 0, stream>>>(bufB, lnG + l * HID,
                                                  lnB + l * HID, lnbf);
    // ff1 = relu(ln @ W1^T + b1): grid (2048/128)*(2048/64) = 512 blocks
    gemm_wmma<EPI_BIAS_RELU, true><<<512, gblk, 0, stream>>>(
        lnbf, wF1b + (size_t)l * FFD * HID, fb1 + l * FFD, nullptr, ffbf,
        MROWS, FFD, HID);
    // out = out + ff1 @ W2^T + b2
    float* dst = (l == LNUM - 1) ? out : bufA;
    gemm_wmma<EPI_BIAS_RES, false><<<128, gblk, 0, stream>>>(
        ffbf, wF2b + (size_t)l * HID * FFD, fb2 + l * HID, bufB, dst,
        MROWS, HID, FFD);
    cur = dst;
  }
}
